// RPN_3444563772181
// MI455X (gfx1250) — matrix operand order repaired and checked by reading, not compiled, not run on hardware
//
#include <hip/hip_runtime.h>
#include <hip/hip_bf16.h>
#include <math.h>

typedef __attribute__((ext_vector_type(16))) _Float16 v16h;
typedef __attribute__((ext_vector_type(8)))  float    v8f;
typedef __attribute__((ext_vector_type(8)))  unsigned int v8u;

#define IMG_H 800.0f
#define IMG_W 1216.0f
#define CH    256
#define FH    200
#define FW    304
#define A_PER 3
#define M_TOT (FH*FW*A_PER)      // 182400 per image
#define SCALE_CLAMP 4.135166556742356f
#define NMS_T 0.7f
#define KTOP  1000
#define CAND  2048

// ---------------------------------------------------------------------------
// Kernel 0: repack w_conv [O=256][I=256][3][3] f32 ->
//   wB[t][cb][o][kh][v]  (t=tap 0..8, cb=k-chunk 0..7, o=out ch 0..255,
//                         kh=lane K-half 0..1, v=VGPR index 0..7)
// u32 element packs f16(w[o][cb*32+kh*16+2v][t]) | f16(...+1)<<16.
// A lane's entire 8-dword B fragment is CONTIGUOUS (32 B) -> 2x b128 loads.
// ---------------------------------------------------------------------------
__global__ void repack_w_kernel(const float* __restrict__ w, unsigned int* __restrict__ wB) {
    int idx = blockIdx.x * blockDim.x + threadIdx.x;
    if (idx >= 9 * 8 * 256 * 2 * 8) return;
    int v  = idx % 8;
    int kh = (idx / 8) % 2;
    int o  = (idx / 16) % 256;
    int cb = (idx / (16 * 256)) % 8;
    int t  = idx / (16 * 256 * 8);
    int i0 = cb * 32 + kh * 16 + 2 * v;
    float w0 = w[((size_t)o * CH + i0)     * 9 + t];
    float w1 = w[((size_t)o * CH + i0 + 1) * 9 + t];
    _Float16 h0 = (_Float16)w0, h1 = (_Float16)w1;
    unsigned short u0 = __builtin_bit_cast(unsigned short, h0);
    unsigned short u1 = __builtin_bit_cast(unsigned short, h1);
    wB[idx] = (unsigned int)u0 | ((unsigned int)u1 << 16);
}

// ---------------------------------------------------------------------------
// Kernel 1: fused conv3x3(relu) + 1x1 obj/delta heads + anchor decode.
// Block = 256 threads (8 wave32s), tile = 32 pixels (one row) x 256 out ch.
// Implicit GEMM: 72 k-steps of V_WMMA_F32_16X16X32_F16 per fragment.
// ---------------------------------------------------------------------------
__global__ void __launch_bounds__(256)
rpn_conv_kernel(const float* __restrict__ feat,
                const unsigned int* __restrict__ wB,
                const float* __restrict__ b_conv,
                const float* __restrict__ w_obj, const float* __restrict__ b_obj,
                const float* __restrict__ w_delta, const float* __restrict__ b_delta,
                float* __restrict__ scores, float* __restrict__ boxes)
{
    // LDS: phase 1 = f16 input tile [3 rows][34 cols][256 ch]  (52224 B)
    //      phase 2 = f32 t tile [32][256] (32768) + head [32][15] + wH[15][256]
    __shared__ __align__(16) unsigned char smem[3 * 34 * CH * 2];
    _Float16* inT   = (_Float16*)smem;
    float*    ldsT  = (float*)smem;                   // [32][256]
    float*    ldsHd = (float*)(smem + 32768);         // [32][15]
    float*    wH    = (float*)(smem + 34816);         // [15][256]

    const int n  = blockIdx.z;
    const int y  = blockIdx.y;
    const int x0 = blockIdx.x * 32;
    const int tid = threadIdx.x;

    // ---- stage input tile to LDS (f32 -> f16), channel-contiguous ----
    const float* f_n = feat + (size_t)n * CH * FH * FW;
    for (int i = tid; i < 768 * 34; i += 256) {       // 768 (c,r) pairs * 34 cols
        int col = i % 34;
        int cr  = i / 34;
        int c   = cr / 3;
        int r   = cr % 3;
        int yy  = y + r - 1;
        int xx  = x0 + col - 1;
        float v = 0.0f;
        if (yy >= 0 && yy < FH && xx >= 0 && xx < FW)
            v = f_n[((size_t)c * FH + yy) * FW + xx];
        inT[((r * 34) + col) * CH + c] = (_Float16)v;
    }
    __syncthreads();

    // ---- implicit GEMM main loop ----
    const int lane = tid & 31;
    const int wv   = tid >> 5;          // 0..7
    const int mh   = wv >> 2;           // pixel half (0..1)
    const int nq   = wv & 3;            // out-channel quarter (0..3)

    const int mrow = lane & 15;
    const int pix  = mh * 16 + mrow;    // A row this lane owns for ds loads
    const int kbA  = (lane < 16) ? 0 : 8;   // A-layout K base
    const int hB   = (lane < 16) ? 0 : 1;   // B-layout K half
    const int ob   = lane & 15;             // B column sub-index

    v8f acc[4];
#pragma unroll
    for (int j = 0; j < 4; ++j)
#pragma unroll
        for (int g = 0; g < 8; ++g) acc[j][g] = 0.0f;

    for (int t = 0; t < 9; ++t) {
        const int r = t / 3, s = t % 3;
        const _Float16* aBase = inT + ((r * 34) + (pix + s)) * CH;
        const unsigned int* bTap = wB + (size_t)t * 8 * 4096;   // 4096 u32 per k-chunk
        for (int cb = 0; cb < 8; ++cb) {
            const int c0 = cb * 32;
            // A fragment: contiguous k = 16h + s order, same as wB (merges to 2x ds_load_b128)
            v8u au;
#pragma unroll
            for (int v = 0; v < 8; ++v) {
                int kp = 2 * kbA + 2 * v;
                au[v] = *(const unsigned int*)(aBase + c0 + kp);
            }
            v16h a = __builtin_bit_cast(v16h, au);

            const unsigned int* bChunk = bTap + (size_t)cb * 4096;
            // prefetch next k-chunk of weights into L2 (global_prefetch_b8)
            __builtin_prefetch((const void*)(bChunk + 4096 + (ob * 2 + hB) * 8), 0, 1);
#pragma unroll
            for (int j = 0; j < 4; ++j) {
                const int o = nq * 64 + j * 16 + ob;
                // contiguous 32-B fragment -> 2x global_load_b128
                v8u bu = *(const v8u*)(bChunk + ((o * 2 + hB) << 3));
                v16h b = __builtin_bit_cast(v16h, bu);
                acc[j] = __builtin_amdgcn_wmma_f32_16x16x32_f16(
                    false, a, false, b, (short)0, acc[j], false, false);
            }
        }
    }
    __syncthreads();   // done reading inT; smem is reused below

    // ---- bias + relu, scatter t-tile to LDS (C/D layout: VGPR g -> M=g+8*(lane>=16)) ----
#pragma unroll
    for (int j = 0; j < 4; ++j) {
        const int o = nq * 64 + j * 16 + (lane & 15);
        const float bc = b_conv[o];
#pragma unroll
        for (int g = 0; g < 8; ++g) {
            const int p = mh * 16 + g + ((lane < 16) ? 0 : 8);
            float v = acc[j][g] + bc;
            ldsT[p * CH + o] = v > 0.0f ? v : 0.0f;
        }
    }
    // stage 1x1 head weights to LDS
    for (int i = tid; i < 15 * CH; i += 256)
        wH[i] = (i < 3 * CH) ? w_obj[i] : w_delta[i - 3 * CH];
    __syncthreads();

    // ---- 1x1 heads: 32 pixels x 15 outputs, dot over 256 channels ----
    for (int id = tid; id < 32 * 15; id += 256) {
        const int p = id / 15, c = id % 15;
        const float bias = (c < 3) ? b_obj[c] : b_delta[c - 3];
        const float* wr = wH + c * CH;
        const float* tr = ldsT + p * CH;
        float s = bias;
#pragma unroll 8
        for (int k = 0; k < CH; ++k) s = fmaf(tr[k], wr[k], s);
        ldsHd[p * 15 + c] = s;
    }
    __syncthreads();

    // ---- anchor decode + clip + score for 32 px x 3 anchors ----
    if (tid < 96) {
        const int p = tid / 3, a = tid % 3;
        const int x = x0 + p;
        if (x < FW) {
            const float logit = ldsHd[p * 15 + a];
            float dx = ldsHd[p * 15 + 3 + a * 4 + 0];
            float dy = ldsHd[p * 15 + 3 + a * 4 + 1];
            float dw = ldsHd[p * 15 + 3 + a * 4 + 2];
            float dh = ldsHd[p * 15 + 3 + a * 4 + 3];
            const float ar = (a == 0) ? 0.5f : ((a == 1) ? 1.0f : 2.0f);
            const float wa = sqrtf(1024.0f / ar);
            const float ha = ar * wa;
            const float cx = (float)x * 4.0f;
            const float cy = (float)y * 4.0f;
            dw = fminf(dw, SCALE_CLAMP);
            dh = fminf(dh, SCALE_CLAMP);
            const float px = dx * wa + cx;
            const float py = dy * ha + cy;
            const float pw = expf(dw) * wa;
            const float ph = expf(dh) * ha;
            float x1 = fminf(fmaxf(px - 0.5f * pw, 0.0f), IMG_W);
            float y1 = fminf(fmaxf(py - 0.5f * ph, 0.0f), IMG_H);
            float x2 = fminf(fmaxf(px + 0.5f * pw, 0.0f), IMG_W);
            float y2 = fminf(fmaxf(py + 0.5f * ph, 0.0f), IMG_H);
            const bool valid = ((x2 - x1) >= 0.0f) && ((y2 - y1) >= 0.0f);
            const size_t idx = (size_t)n * M_TOT + ((size_t)y * FW + x) * A_PER + a;
            scores[idx] = valid ? logit : -INFINITY;
            float* bxp = boxes + idx * 4;
            bxp[0] = x1; bxp[1] = y1; bxp[2] = x2; bxp[3] = y2;
        }
    }
}

// ---------------------------------------------------------------------------
// Kernel 2: per-image exact top-1000 (radix-select + bitonic sort) + NMS +
// stable partition (== final top_k over {score, -inf}).
// ---------------------------------------------------------------------------
__device__ __forceinline__ unsigned int f2k(float f) {
    unsigned int u = __float_as_uint(f);
    return (u & 0x80000000u) ? ~u : (u | 0x80000000u);
}
__device__ __forceinline__ float k2f(unsigned int k) {
    unsigned int u = (k & 0x80000000u) ? (k & 0x7FFFFFFFu) : ~k;
    return __uint_as_float(u);
}

__global__ void __launch_bounds__(1024)
topk_nms_kernel(const float* __restrict__ scores, const float* __restrict__ boxes,
                float* __restrict__ out_b, float* __restrict__ out_s)
{
    const int n = blockIdx.x;
    const int tid = threadIdx.x;
    const float* sc = scores + (size_t)n * M_TOT;
    const float* bx = boxes  + (size_t)n * M_TOT * 4;

    __shared__ unsigned int hist[256];
    __shared__ unsigned int s_pfx;
    __shared__ int s_k;
    __shared__ int s_cc;
    __shared__ unsigned long long cand[CAND];
    __shared__ float nb[KTOP * 4];
    __shared__ float ns[KTOP];
    __shared__ unsigned char sup[KTOP];
    __shared__ int pos[KTOP];
    __shared__ int s_nkeep;

    // ---- radix-select the exact 1000th-largest key ----
    if (tid == 0) { s_pfx = 0u; s_k = KTOP; }
    __syncthreads();
    for (int pass = 3; pass >= 0; --pass) {
        for (int i = tid; i < 256; i += 1024) hist[i] = 0u;
        __syncthreads();
        const unsigned int pmask = (pass == 3) ? 0u : (0xFFFFFFFFu << ((pass + 1) * 8));
        const unsigned int pfxv = s_pfx;
        for (int i = tid; i < M_TOT; i += 1024) {
            unsigned int key = f2k(sc[i]);
            if ((key & pmask) == pfxv)
                atomicAdd(&hist[(key >> (pass * 8)) & 0xFF], 1u);
        }
        __syncthreads();
        if (tid == 0) {
            int k = s_k;
            int b = 255;
            for (; b > 0; --b) {
                int c = (int)hist[b];
                if (k - c <= 0) break;
                k -= c;
            }
            s_pfx = pfxv | ((unsigned int)b << (pass * 8));
            s_k = k;
        }
        __syncthreads();
    }
    const unsigned int T = s_pfx;

    // ---- gather candidates (key >= T) as (key | ~idx) composites ----
    if (tid == 0) s_cc = 0;
    __syncthreads();
    for (int i = tid; i < M_TOT; i += 1024) {
        unsigned int key = f2k(sc[i]);
        if (key >= T) {
            int p = atomicAdd(&s_cc, 1);
            if (p < CAND)
                cand[p] = ((unsigned long long)key << 32) |
                          (unsigned long long)(0xFFFFFFFFu - (unsigned int)i);
        }
    }
    __syncthreads();
    const int cn = s_cc < CAND ? s_cc : CAND;
    for (int i = cn + tid; i < CAND; i += 1024) cand[i] = 0ull;
    __syncthreads();

    // ---- bitonic sort descending, 1024 comparators per stage ----
    for (int size = 2; size <= CAND; size <<= 1) {
        for (int stride = size >> 1; stride > 0; stride >>= 1) {
            const int lo = 2 * stride * (tid / stride) + (tid % stride);
            const int hi = lo + stride;
            const bool desc = ((lo & size) == 0);
            unsigned long long a = cand[lo], b = cand[hi];
            const bool sw = desc ? (a < b) : (a > b);
            if (sw) { cand[lo] = b; cand[hi] = a; }
            __syncthreads();
        }
    }

    // ---- materialize top-1000 boxes/scores ----
    for (int k = tid; k < KTOP; k += 1024) {
        const unsigned long long c = cand[k];
        const unsigned int key = (unsigned int)(c >> 32);
        const unsigned int idx = 0xFFFFFFFFu - (unsigned int)(c & 0xFFFFFFFFull);
        ns[k] = k2f(key);
        const float* b = bx + (size_t)idx * 4;
        nb[k * 4 + 0] = b[0]; nb[k * 4 + 1] = b[1];
        nb[k * 4 + 2] = b[2]; nb[k * 4 + 3] = b[3];
        sup[k] = 0;
    }
    __syncthreads();

    // ---- greedy NMS (matches reference scan semantics) ----
    for (int i = 0; i < KTOP; ++i) {
        const bool keepi = (sup[i] == 0);
        const float ax1 = nb[i * 4 + 0], ay1 = nb[i * 4 + 1];
        const float ax2 = nb[i * 4 + 2], ay2 = nb[i * 4 + 3];
        if (keepi && tid < KTOP && tid > i && sup[tid] == 0) {
            const int j = tid;
            const float bx1 = nb[j * 4 + 0], by1 = nb[j * 4 + 1];
            const float bx2 = nb[j * 4 + 2], by2 = nb[j * 4 + 3];
            const float areaA = (ax2 - ax1) * (ay2 - ay1);
            const float areaB = (bx2 - bx1) * (by2 - by1);
            const float ix1 = fmaxf(ax1, bx1), iy1 = fmaxf(ay1, by1);
            const float ix2 = fminf(ax2, bx2), iy2 = fminf(ay2, by2);
            const float iw = fmaxf(ix2 - ix1, 0.0f);
            const float ih = fmaxf(iy2 - iy1, 0.0f);
            const float inter = iw * ih;
            const float iou = inter / fmaxf(areaA + areaB - inter, 1e-9f);
            if (iou > NMS_T) sup[j] = 1;
        }
        __syncthreads();
    }

    // ---- stable partition == final top_k over {score, -inf} ----
    if (tid == 0) {
        int c = 0;
        for (int k = 0; k < KTOP; ++k) if (!sup[k]) pos[k] = c++;
        s_nkeep = c;
        for (int k = 0; k < KTOP; ++k) if (sup[k]) pos[k] = c++;
    }
    __syncthreads();

    float* ob = out_b + (size_t)n * KTOP * 4;
    float* os = out_s + (size_t)n * KTOP;
    for (int k = tid; k < KTOP; k += 1024) {
        const int p = pos[k];
        const bool kp = (sup[k] == 0);
        os[p] = kp ? ns[k] : -INFINITY;
        ob[p * 4 + 0] = kp ? nb[k * 4 + 0] : 0.0f;
        ob[p * 4 + 1] = kp ? nb[k * 4 + 1] : 0.0f;
        ob[p * 4 + 2] = kp ? nb[k * 4 + 2] : 0.0f;
        ob[p * 4 + 3] = kp ? nb[k * 4 + 3] : 0.0f;
    }
}

// ---------------------------------------------------------------------------
extern "C" void kernel_launch(void* const* d_in, const int* in_sizes, int n_in,
                              void* d_out, int out_size, void* d_ws, size_t ws_size,
                              hipStream_t stream) {
    const float* feat    = (const float*)d_in[0];
    const float* w_conv  = (const float*)d_in[1];
    const float* b_conv  = (const float*)d_in[2];
    const float* w_obj   = (const float*)d_in[3];
    const float* b_obj   = (const float*)d_in[4];
    const float* w_delta = (const float*)d_in[5];
    const float* b_delta = (const float*)d_in[6];

    unsigned char* ws = (unsigned char*)d_ws;
    unsigned int* wB = (unsigned int*)ws;                        // 1,179,648 B
    float* scores = (float*)(ws + 1179648);                      // 1,459,200 B
    float* boxes  = (float*)(ws + 1179648 + 1459200);            // 5,836,800 B

    // 1) repack conv weights to f16 WMMA-B fragment-contiguous layout
    repack_w_kernel<<<(9 * 8 * 256 * 2 * 8 + 255) / 256, 256, 0, stream>>>(w_conv, wB);

    // 2) fused conv + heads + decode
    dim3 grid(10, FH, 2);   // 10 x-tiles of 32 px, 200 rows, 2 images
    rpn_conv_kernel<<<grid, 256, 0, stream>>>(feat, wB, b_conv, w_obj, b_obj,
                                              w_delta, b_delta, scores, boxes);

    // 3) per-image top-k + NMS + final ordering
    float* out_b = (float*)d_out;            // [2][1000][4]
    float* out_s = out_b + 2 * KTOP * 4;     // [2][1000]
    topk_nms_kernel<<<2, 1024, 0, stream>>>(scores, boxes, out_b, out_s);
}
